// BaseAttentionBlock_29317446762621
// MI455X (gfx1250) — compile-verified, matrix operand order checked by reading, not run on hardware
//
#include <hip/hip_runtime.h>
#include <hip/hip_bf16.h>

// ---------------------------------------------------------------------------
// Problem constants
// ---------------------------------------------------------------------------
#define D_MODEL 1024
#define NUM_HEADS 16
#define D_K 64
#define BATCH 4
#define SEQ 2048
#define MTOK (BATCH * SEQ)   // 8192 tokens

typedef __attribute__((ext_vector_type(16))) __bf16        v16bf;
typedef __attribute__((ext_vector_type(8)))  float         v8f;
typedef __attribute__((ext_vector_type(4)))  unsigned int  u32x4;
typedef __attribute__((ext_vector_type(8)))  int           i32x8;
typedef __attribute__((ext_vector_type(4)))  int           i32x4;

union Frag16 {
    uint4 u[2];
    v16bf v;
};

// --------------------------- feature detection ------------------------------
#if defined(__has_builtin)
#if __has_builtin(__builtin_amdgcn_tensor_load_to_lds)
#define HAVE_TDM 1
#endif
#if __has_builtin(__builtin_amdgcn_s_wait_tensorcnt)
#define HAVE_TENSORCNT_BUILTIN 1
#endif
#endif
#ifndef HAVE_TDM
#define HAVE_TDM 0
#endif
#ifndef HAVE_TENSORCNT_BUILTIN
#define HAVE_TENSORCNT_BUILTIN 0
#endif

__device__ __forceinline__ void wait_tensorcnt0() {
#if HAVE_TENSORCNT_BUILTIN
    __builtin_amdgcn_s_wait_tensorcnt(0);
#else
    asm volatile("s_wait_tensorcnt 0" ::: "memory");
#endif
}

// ---------------------------------------------------------------------------
// Fragment load: 16x32 bf16 A/B fragment from row-major matrix (ld elems/row).
// lane L (r=L&15, h=L>>4) holds row r, K-chunks [8h..8h+7], [16+8h..16+8h+7].
// ---------------------------------------------------------------------------
__device__ __forceinline__ v16bf load_frag(const __bf16* __restrict__ p, int ld, int lane) {
    const int r  = lane & 15;
    const int hh = lane >> 4;
    const __bf16* q = p + (size_t)r * ld + hh * 8;
    Frag16 f;
    f.u[0] = *(const uint4*)(q);
    f.u[1] = *(const uint4*)(q + 16);
    return f.v;
}

// Same fragment shape out of an LDS tile laid out row-major [rows][32] bf16.
__device__ __forceinline__ v16bf load_frag_lds32(const __bf16* p, int row0, int lane) {
    const int r  = lane & 15;
    const int hh = lane >> 4;
    const __bf16* q = p + (row0 + r) * 32 + hh * 8;
    Frag16 f;
    f.u[0] = *(const uint4*)(q);
    f.u[1] = *(const uint4*)(q + 16);
    return f.v;
}

__device__ __forceinline__ v8f wmma_bf16(v16bf a, v16bf b, v8f c) {
    return __builtin_amdgcn_wmma_f32_16x16x32_bf16(
        false, a, false, b, (short)0, c, false, false);
}

// Max-reduction across the 16 lanes of one half-wave (masks < 16 stay in-half).
__device__ __forceinline__ float hmax16(float v) {
    v = fmaxf(v, __shfl_xor(v, 1));
    v = fmaxf(v, __shfl_xor(v, 2));
    v = fmaxf(v, __shfl_xor(v, 4));
    v = fmaxf(v, __shfl_xor(v, 8));
    return v;
}

// ---------------------------------------------------------------------------
// fp32 -> bf16 conversion (grid-stride)
// ---------------------------------------------------------------------------
__global__ __launch_bounds__(256) void cvt_f32_bf16(const float* __restrict__ in,
                                                    __bf16* __restrict__ out, int n) {
    for (int i = blockIdx.x * blockDim.x + threadIdx.x; i < n; i += gridDim.x * blockDim.x)
        out[i] = (__bf16)in[i];
}

// ---------------------------------------------------------------------------
// TDM: 2D tile load (128 rows x 64 bytes) -> LDS, descriptor per ISA 8.3/8.4.
//   data_size = 4B units; tile_dim0 = 16 units (32 bf16), tile_dim1 = 128 rows;
//   tensor_dim0_stride = 512 units (1024 bf16 row).
// ---------------------------------------------------------------------------
#if HAVE_TDM
__device__ __forceinline__ void tdm_load_tile_128x32(const __bf16* gsrc, unsigned lds_off) {
    const unsigned long long ga = (unsigned long long)(uintptr_t)gsrc;
    u32x4 g0;
    g0[0] = 1u;                                    // count=1 (valid), no gather
    g0[1] = lds_off;                               // LDS byte address
    g0[2] = (unsigned)ga;                          // global_addr[31:0]
    g0[3] = (unsigned)(ga >> 32) | (2u << 30);     // global_addr[56:32] | type=2
    i32x8 g1;
    g1[0] = (int)(2u << 16);                       // wg_mask=0, data_size=4B
    g1[1] = (int)((512u & 0xFFFFu) << 16);         // tensor_dim0[15:0]
    g1[2] = (int)((512u >> 16) | (8192u << 16));   // tensor_dim0[31:16] | dim1[15:0]
    g1[3] = (int)((8192u >> 16) | (16u << 16));    // dim1[31:16] | tile_dim0=16
    g1[4] = (int)128u;                             // tile_dim1=128, tile_dim2=0 (2D)
    g1[5] = (int)512;                              // tensor_dim0_stride[31:0]
    g1[6] = 0;                                     // stride0[47:32] | stride1[15:0]
    g1[7] = 0;
    i32x4 z4 = {0, 0, 0, 0};
#if __clang_major__ >= 23
    i32x8 z8 = {0, 0, 0, 0, 0, 0, 0, 0};
    __builtin_amdgcn_tensor_load_to_lds(g0, g1, z4, z4, z8, 0);
#else
    __builtin_amdgcn_tensor_load_to_lds(g0, g1, z4, z4, 0);
#endif
}
#endif

// Issue staging of A-tile(128x32) and W-tile(128x32) into one LDS buffer.
// TDM path: wave 0 issues async DMA (completion via wait_tensorcnt0 later).
// Fallback: cooperative loads (completion via the workgroup barrier).
__device__ __forceinline__ void stage_issue(const __bf16* __restrict__ Atile,
                                            const __bf16* __restrict__ Wtile,
                                            __bf16* lA, __bf16* lB) {
#if HAVE_TDM
    if (threadIdx.x < 32) {                         // one wave issues the TDM ops
        tdm_load_tile_128x32(Atile, (unsigned)(uintptr_t)lA);
        tdm_load_tile_128x32(Wtile, (unsigned)(uintptr_t)lB);
    }
#else
    uint4* dA = (uint4*)lA;
    uint4* dB = (uint4*)lB;
    for (int i = threadIdx.x; i < 512; i += 256) {  // 128 rows x 4 quads
        const int row = i >> 2, qd = i & 3;
        dA[i] = *(const uint4*)(Atile + (size_t)row * D_MODEL + qd * 8);
        dB[i] = *(const uint4*)(Wtile + (size_t)row * D_MODEL + qd * 8);
    }
#endif
}

__device__ __forceinline__ void stage_finish() {
#if HAVE_TDM
    if (threadIdx.x < 32) wait_tensorcnt0();
#endif
}

// ---------------------------------------------------------------------------
// GEMM:  Y[m,n] = (sum_k A[m,k]*W[n,k] + bias[n]) * oscale
//   Block 256 thr = 8 waves, block tile 128(M) x 128(N), wave tile 64 x 32.
//   K-step tiles staged into double-buffered LDS via TDM (DMA overlaps WMMA).
//   MODE 0: bf16 row-major   MODE 1: bf16 per-head-transposed V   MODE 2: f32
// ---------------------------------------------------------------------------
template <int MODE>
__global__ __launch_bounds__(256) void gemm_bias(const __bf16* __restrict__ A,
                                                 const __bf16* __restrict__ W,
                                                 const float* __restrict__ bias,
                                                 void* __restrict__ out,
                                                 float oscale) {
    __shared__ __bf16 lA[2][128 * 32];   // 2 x 8 KB
    __shared__ __bf16 lB[2][128 * 32];   // 2 x 8 KB

    const int K = D_MODEL, N = D_MODEL;
    const int wave = threadIdx.x >> 5;
    const int lane = threadIdx.x & 31;
    const int r  = lane & 15;
    const int hh = lane >> 4;

    const int mw = wave & 1;          // 2 waves along M
    const int nw = wave >> 1;         // 4 waves along N
    const int mblk = blockIdx.y * 128;
    const int nblk = blockIdx.x * 128;
    const int m0 = mblk + mw * 64;
    const int n0 = nblk + nw * 32;

    const __bf16* Abase = A + (size_t)mblk * K;
    const __bf16* Wbase = W + (size_t)nblk * K;

    v8f acc[4][2] = {};

    const int NSTEP = K / 32;         // 32 K-steps

    // prologue: stage step 0 into buffer 0
    stage_issue(Abase, Wbase, lA[0], lB[0]);
    stage_finish();
    __syncthreads();

    for (int s = 0; s < NSTEP; ++s) {
        const int cur = s & 1;
        const int nxt = cur ^ 1;
        if (s + 1 < NSTEP)            // overlap next DMA with this step's WMMAs
            stage_issue(Abase + (s + 1) * 32, Wbase + (s + 1) * 32, lA[nxt], lB[nxt]);

        v16bf bfr0 = load_frag_lds32(lB[cur], nw * 32 + 0,  lane);
        v16bf bfr1 = load_frag_lds32(lB[cur], nw * 32 + 16, lane);
#pragma unroll
        for (int t = 0; t < 4; ++t) {
            v16bf afr = load_frag_lds32(lA[cur], mw * 64 + t * 16, lane);
            acc[t][0] = wmma_bf16(afr, bfr0, acc[t][0]);
            acc[t][1] = wmma_bf16(afr, bfr1, acc[t][1]);
        }

        stage_finish();               // TDM for step s+1 must be done
        __syncthreads();              // everyone done reading cur, nxt ready
    }

#pragma unroll
    for (int t = 0; t < 2; ++t) {
        const int   col = n0 + t * 16 + r;
        const float bv  = bias[col];
#pragma unroll
        for (int s = 0; s < 4; ++s) {
            const int mbase = m0 + s * 16 + hh * 8;
#pragma unroll
            for (int rr = 0; rr < 8; ++rr) {
                const float val = (acc[s][t][rr] + bv) * oscale;
                const int m = mbase + rr;
                if (MODE == 0) {
                    ((__bf16*)out)[(size_t)m * N + col] = (__bf16)val;
                } else if (MODE == 1) {
                    const int bb = m >> 11, ss = m & 2047;
                    const int h = col >> 6,  c = col & 63;
                    ((__bf16*)out)[((size_t)((bb * NUM_HEADS + h) * D_K + c)) * SEQ + ss] =
                        (__bf16)val;
                } else {
                    ((float*)out)[(size_t)m * N + col] = val;
                }
            }
        }
    }
}

// ---------------------------------------------------------------------------
// One 32-key flash-attention tile.  Q arrives pre-scaled by 1/sqrt(D_K).
// Row-sums of P are computed with an extra WMMA against a ones-matrix
// (replaces a 32-bpermute butterfly per tile and matches bf16-P numerics).
// ---------------------------------------------------------------------------
template <bool MASK>
__device__ __forceinline__ void attn_tile(int j0, int q0, int lane, int r, int hh,
                                          const __bf16* __restrict__ Kbase,
                                          const __bf16* __restrict__ Vbase,
                                          __bf16* Pl, v16bf onesf,
                                          v16bf qf0, v16bf qf1,
                                          v8f (&ctx)[4], float (&mrow)[8],
                                          float (&lrow)[8]) {
    // scores: S(16x32) = Qs(16x64) @ K^T(64x32)
    v8f sc0 = {}, sc1 = {};
    {
        v16bf kf = load_frag(Kbase + (size_t)(j0 + 0)  * D_MODEL + 0,  D_MODEL, lane);
        sc0 = wmma_bf16(qf0, kf, sc0);
        kf  = load_frag(Kbase + (size_t)(j0 + 0)  * D_MODEL + 32, D_MODEL, lane);
        sc0 = wmma_bf16(qf1, kf, sc0);
        kf  = load_frag(Kbase + (size_t)(j0 + 16) * D_MODEL + 0,  D_MODEL, lane);
        sc1 = wmma_bf16(qf0, kf, sc1);
        kf  = load_frag(Kbase + (size_t)(j0 + 16) * D_MODEL + 32, D_MODEL, lane);
        sc1 = wmma_bf16(qf1, kf, sc1);
    }

    const int qrow_base = q0 + 8 * hh;
#pragma unroll
    for (int rr = 0; rr < 8; ++rr) {
        float v0 = sc0[rr];
        float v1 = sc1[rr];
        if (MASK) {
            const int qg = qrow_base + rr;
            if (j0 + r      > qg) v0 = -3.0e38f;
            if (j0 + 16 + r > qg) v1 = -3.0e38f;
        }
        float tmax = hmax16(fmaxf(v0, v1));
        const float mn = fmaxf(mrow[rr], tmax);
        const float p0 = __expf(v0 - mn);
        const float p1 = __expf(v1 - mn);
        const float alpha = __expf(mrow[rr] - mn);
        mrow[rr] = mn;
        lrow[rr] *= alpha;
        ctx[0][rr] *= alpha;
        ctx[1][rr] *= alpha;
        ctx[2][rr] *= alpha;
        ctx[3][rr] *= alpha;

        Pl[(rr + 8 * hh) * 32 + r]      = (__bf16)p0;   // C-layout -> row-major LDS
        Pl[(rr + 8 * hh) * 32 + 16 + r] = (__bf16)p1;
    }

    // same-wave LDS ordering (DS pipe is in-order; fence compiler + HW)
    asm volatile("s_wait_dscnt 0" ::: "memory");

    Frag16 pf;
    {
        const __bf16* pp = &Pl[r * 32 + hh * 8];
        pf.u[0] = *(const uint4*)(pp);
        pf.u[1] = *(const uint4*)(pp + 16);
    }

    // row sums via matrix unit: P(16x32) @ ones(32x16) -> rowsum in C-layout
    v8f zc = {};
    v8f ps = wmma_bf16(pf.v, onesf, zc);
#pragma unroll
    for (int rr = 0; rr < 8; ++rr) lrow[rr] += ps[rr];

    // ctx(16x64) += P(16x32) @ V(32x64)  (B-frags from per-head transposed V)
#pragma unroll
    for (int nf = 0; nf < 4; ++nf) {
        v16bf vf = load_frag(Vbase + (size_t)(nf * 16) * SEQ + j0, SEQ, lane);
        ctx[nf] = wmma_bf16(pf.v, vf, ctx[nf]);
    }
}

// ---------------------------------------------------------------------------
// Flash attention (causal), one wave per (b, h, 16-query tile).
// ---------------------------------------------------------------------------
__global__ __launch_bounds__(128) void attn_causal(const __bf16* __restrict__ Qb,
                                                   const __bf16* __restrict__ Kb,
                                                   const __bf16* __restrict__ Vt,
                                                   __bf16* __restrict__ Cb) {
    __shared__ __bf16 Plds[4][16 * 32];

    const int winb = threadIdx.x >> 5;
    const int lane = threadIdx.x & 31;
    const int r  = lane & 15;
    const int hh = lane >> 4;

    const int gw = blockIdx.x * 4 + winb;    // 0 .. 8191
    const int qt = gw & 127;
    const int bh = gw >> 7;
    const int b  = bh >> 4;
    const int h  = bh & 15;
    const int q0 = qt * 16;

    const __bf16* Qbase = Qb + (size_t)(b * SEQ) * D_MODEL + h * D_K;
    const __bf16* Kbase = Kb + (size_t)(b * SEQ) * D_MODEL + h * D_K;
    const __bf16* Vbase = Vt + (size_t)bh * D_K * SEQ;
    __bf16* Pl = &Plds[winb][0];

    // Resident pre-scaled Q fragments (16x64 = two 16x32 k-steps)
    v16bf qf0 = load_frag(Qbase + (size_t)q0 * D_MODEL + 0,  D_MODEL, lane);
    v16bf qf1 = load_frag(Qbase + (size_t)q0 * D_MODEL + 32, D_MODEL, lane);

    v16bf onesf;
#pragma unroll
    for (int i = 0; i < 16; ++i) onesf[i] = (__bf16)1.0f;

    v8f ctx[4] = {};
    float mrow[8], lrow[8];
#pragma unroll
    for (int i = 0; i < 8; ++i) { mrow[i] = -3.0e38f; lrow[i] = 0.0f; }

    // full (mask-free) tiles: j0 + 31 <= q0
    const int nfull = (q0 + 1) >> 5;
    int j0 = 0;
    for (int t = 0; t < nfull; ++t, j0 += 32) {
        __builtin_prefetch(Kbase + (size_t)(j0 + 32) * D_MODEL, 0, 1);
        __builtin_prefetch(Vbase + (size_t)(j0 + 32), 0, 1);
        attn_tile<false>(j0, q0, lane, r, hh, Kbase, Vbase, Pl, onesf, qf0, qf1,
                         ctx, mrow, lrow);
    }
    // 1-2 masked diagonal tiles
    for (; j0 <= q0 + 15; j0 += 32) {
        attn_tile<true>(j0, q0, lane, r, hh, Kbase, Vbase, Pl, onesf, qf0, qf1,
                        ctx, mrow, lrow);
    }

    // normalize and store context
#pragma unroll
    for (int rr = 0; rr < 8; ++rr) {
        const float inv = 1.0f / lrow[rr];
        const int m = q0 + rr + 8 * hh;
        __bf16* crow = Cb + (size_t)(b * SEQ + m) * D_MODEL + h * D_K;
        crow[0 * 16 + r] = (__bf16)(ctx[0][rr] * inv);
        crow[1 * 16 + r] = (__bf16)(ctx[1][rr] * inv);
        crow[2 * 16 + r] = (__bf16)(ctx[2][rr] * inv);
        crow[3 * 16 + r] = (__bf16)(ctx[3][rr] * inv);
    }
}

// ---------------------------------------------------------------------------
// Host side
// ---------------------------------------------------------------------------
extern "C" void kernel_launch(void* const* d_in, const int* in_sizes, int n_in,
                              void* d_out, int out_size, void* d_ws, size_t ws_size,
                              hipStream_t stream) {
    const float* x  = (const float*)d_in[0];
    const float* Wq = (const float*)d_in[1];
    const float* bq = (const float*)d_in[2];
    const float* Wk = (const float*)d_in[3];
    const float* bk = (const float*)d_in[4];
    const float* Wv = (const float*)d_in[5];
    const float* bv = (const float*)d_in[6];
    const float* Wo = (const float*)d_in[7];
    const float* bo = (const float*)d_in[8];

    const size_t MB = 1024ull * 1024ull;
    char* ws = (char*)d_ws;
    __bf16* Xb  = (__bf16*)(ws + 0 * MB);     // 16 MB
    __bf16* Wqb = (__bf16*)(ws + 16 * MB);    //  2 MB
    __bf16* Wkb = (__bf16*)(ws + 18 * MB);    //  2 MB
    __bf16* Wvb = (__bf16*)(ws + 20 * MB);    //  2 MB
    __bf16* Wob = (__bf16*)(ws + 22 * MB);    //  2 MB
    __bf16* Qb  = (__bf16*)(ws + 24 * MB);    // 16 MB (pre-scaled by 1/8)
    __bf16* Kb  = (__bf16*)(ws + 40 * MB);    // 16 MB
    __bf16* Vt  = (__bf16*)(ws + 56 * MB);    // 16 MB (per-head transposed V)
    __bf16* Cb  = (__bf16*)(ws + 72 * MB);    // 16 MB   -> total 88 MB

    // 1) convert activations + weights to bf16
    const int NX = MTOK * D_MODEL;
    const int NW = D_MODEL * D_MODEL;
    cvt_f32_bf16<<<4096, 256, 0, stream>>>(x,  Xb,  NX);
    cvt_f32_bf16<<<1024, 256, 0, stream>>>(Wq, Wqb, NW);
    cvt_f32_bf16<<<1024, 256, 0, stream>>>(Wk, Wkb, NW);
    cvt_f32_bf16<<<1024, 256, 0, stream>>>(Wv, Wvb, NW);
    cvt_f32_bf16<<<1024, 256, 0, stream>>>(Wo, Wob, NW);

    // 2) Q/K/V projections (Q pre-scaled by softmax scale; V stored transposed)
    dim3 ggrid(D_MODEL / 128, MTOK / 128);    // 8 x 64
    gemm_bias<0><<<ggrid, 256, 0, stream>>>(Xb, Wqb, bq, (void*)Qb, 0.125f);
    gemm_bias<0><<<ggrid, 256, 0, stream>>>(Xb, Wkb, bk, (void*)Kb, 1.0f);
    gemm_bias<1><<<ggrid, 256, 0, stream>>>(Xb, Wvb, bv, (void*)Vt, 1.0f);

    // 3) causal flash attention: 8192 waves, 4 waves / block
    attn_causal<<<2048, 128, 0, stream>>>(Qb, Kb, Vt, Cb);

    // 4) output projection -> fp32
    gemm_bias<2><<<ggrid, 256, 0, stream>>>(Cb, Wob, bo, d_out, 1.0f);

    (void)in_sizes; (void)n_in; (void)out_size; (void)ws_size;
}
